// ManualTransformerLayer_36936718746009
// MI455X (gfx1250) — matrix operand order repaired and checked by reading, not compiled, not run on hardware
//
#include <hip/hip_runtime.h>
#include <hip/hip_bf16.h>

// ---------------------------------------------------------------------------
// Types for CDNA5 WMMA (wave32): 16x16x32 bf16 -> f32
// ---------------------------------------------------------------------------
typedef __attribute__((ext_vector_type(16))) __bf16 v16bf;
typedef __attribute__((ext_vector_type(8)))  float  v8f;
typedef __attribute__((ext_vector_type(4)))  unsigned int u32x4;

union Frag {
    v16bf v;
    u32x4 q[2];
    unsigned short u[16];
};

__device__ __forceinline__ unsigned short f32_to_bf16(float f) {
    union { float f; unsigned int u; } c; c.f = f;
    unsigned int r = c.u + 0x7FFFu + ((c.u >> 16) & 1u);  // round-to-nearest-even
    return (unsigned short)(r >> 16);
}

// Async global->LDS copy of 16 bytes (GLOBAL_LOAD_ASYNC_TO_LDS_B128, ASYNCcnt).
// ISA 08_async_tensor: the instruction offset is added to both the LDS and global address.
__device__ __forceinline__ void async_copy_b128(void* lds_dst, const void* gsrc,
                                                int ioffset) {
    unsigned lds_off = (unsigned)(unsigned long long)lds_dst;
    unsigned long long ga = (unsigned long long)gsrc;
    if (ioffset == 0)
        asm volatile("global_load_async_to_lds_b128 %0, %1, off"
                     :: "v"(lds_off), "v"(ga) : "memory");
    else
        asm volatile("global_load_async_to_lds_b128 %0, %1, off offset:16"
                     :: "v"(lds_off), "v"(ga) : "memory");
}

__device__ __forceinline__ void wait_async_zero() {
    asm volatile("s_wait_asynccnt 0" ::: "memory");
}

// ---------------------------------------------------------------------------
// fp32 -> bf16 elementwise conversion (weights)
// ---------------------------------------------------------------------------
__global__ __launch_bounds__(256) void cvt_bf16_kernel(
    const float* __restrict__ in, unsigned short* __restrict__ out, int n) {
    int i = blockIdx.x * 256 + threadIdx.x;
    if (i < n) out[i] = f32_to_bf16(in[i]);
}

// ---------------------------------------------------------------------------
// LayerNorm: one row (D floats) per block of 256 threads, bf16 output
// ---------------------------------------------------------------------------
__global__ __launch_bounds__(256) void layernorm_bf16_kernel(
    const float* __restrict__ x, const float* __restrict__ gamma,
    const float* __restrict__ beta, unsigned short* __restrict__ out, int D) {
    __shared__ float red[256];
    const int row = blockIdx.x;
    const float* xr = x + (size_t)row * D;

    float s = 0.f;
    for (int i = threadIdx.x; i < D; i += 256) s += xr[i];
    red[threadIdx.x] = s;
    __syncthreads();
    for (int off = 128; off > 0; off >>= 1) {
        if (threadIdx.x < off) red[threadIdx.x] += red[threadIdx.x + off];
        __syncthreads();
    }
    const float mean = red[0] / (float)D;
    __syncthreads();

    float vs = 0.f;
    for (int i = threadIdx.x; i < D; i += 256) {
        float d = xr[i] - mean;
        vs += d * d;
    }
    red[threadIdx.x] = vs;
    __syncthreads();
    for (int off = 128; off > 0; off >>= 1) {
        if (threadIdx.x < off) red[threadIdx.x] += red[threadIdx.x + off];
        __syncthreads();
    }
    const float rstd = rsqrtf(red[0] / (float)D + 1e-5f);

    for (int i = threadIdx.x; i < D; i += 256) {
        float y = gamma[i] * (xr[i] - mean) * rstd + beta[i];
        out[(size_t)row * D + i] = f32_to_bf16(y);
    }
}

// ---------------------------------------------------------------------------
// Tiled bf16 GEMM with f32 accumulate via v_wmma_f32_16x16x32_bf16.
//   C[M,N] = A[M,K] * B[K,N]  (+bias) (ReLU) (+residual)
// Block: 256 threads = 8 waves. Block tile 128x128, k-step 32.
// Wave w: rows (w/2)*32..+31, cols (w%2)*64..+63 -> 2x4 fragments of 16x16.
// A tile staged with async global->LDS DMA (ASYNCcnt); B tile staged with a
// manual transpose so B fragments are contiguous 16B LDS loads.
// ---------------------------------------------------------------------------
template <bool BIAS, bool RELU, bool RES, bool OUTBF16>
__global__ __launch_bounds__(256) void gemm_bf16_kernel(
    const unsigned short* __restrict__ A, const unsigned short* __restrict__ B,
    const float* __restrict__ bias, const float* __restrict__ resid,
    void* __restrict__ Cout, int M, int N, int K) {
    __shared__ unsigned short As[128 * 32];
    __shared__ unsigned short Bs[128 * 32];

    const int tid   = threadIdx.x;
    const int lane  = tid & 31;
    const int wave  = tid >> 5;
    const int l15   = lane & 15;
    const int lhalf = (lane < 16) ? 0 : 1;

    const int bm = blockIdx.y * 128;
    const int bn = blockIdx.x * 128;
    const int wm = (wave >> 1) * 32;
    const int wn = (wave & 1) * 64;

    const v8f zero = {0.f, 0.f, 0.f, 0.f, 0.f, 0.f, 0.f, 0.f};
    v8f acc[2][4];
    for (int i = 0; i < 2; ++i)
        for (int j = 0; j < 4; ++j) acc[i][j] = zero;

    const int a_row = tid >> 1;            // 0..127
    const int a_col = (tid & 1) * 16;      // 0 or 16
    const int b_kk  = tid & 31;            // 0..31
    const int b_n0  = (tid >> 5) * 16;     // 0..112

    for (int k0 = 0; k0 < K; k0 += 32) {
        // --- stage A tile (128x32 bf16) via async global->LDS DMA ---
        {
            const unsigned short* srcA =
                A + (size_t)(bm + a_row) * K + k0 + a_col;
            unsigned short* dstA = As + a_row * 32 + a_col;
            async_copy_b128(dstA, srcA, 0);
            async_copy_b128(dstA, srcA, 16);
        }
        // --- stage B tile transposed: Bs[n][k] = B[k0+k][bn+n] ---
        {
            const unsigned short* src = B + (size_t)(k0 + b_kk) * N + bn + b_n0;
            #pragma unroll
            for (int j = 0; j < 16; ++j)
                Bs[(b_n0 + j) * 32 + b_kk] = src[j];
            if (k0 + 32 < K)  // prefetch next k-tile (global_prefetch_b8)
                __builtin_prefetch(
                    (const void*)(B + (size_t)(k0 + 32 + b_kk) * N + bn + b_n0),
                    0, 1);
        }
        wait_async_zero();   // wave's async tile DMA complete
        __syncthreads();     // all waves' staging visible

        // --- A fragments (16x32, contiguous K = 16*half + s, same order as B) ---
        Frag a[2];
        #pragma unroll
        for (int i = 0; i < 2; ++i) {
            const u32x4* p = (const u32x4*)(As + (wm + i * 16 + l15) * 32);
            a[i].q[0] = p[lhalf * 2];      // K = {0..7} or {16..23}
            a[i].q[1] = p[lhalf * 2 + 1];  // K = {8..15} or {24..31}
        }
        // --- B fragments (32x16): lane = column, contiguous 16 K values ---
        Frag b[4];
        #pragma unroll
        for (int j = 0; j < 4; ++j) {
            const u32x4* p =
                (const u32x4*)(Bs + (wn + j * 16 + l15) * 32 + lhalf * 16);
            b[j].q[0] = p[0];
            b[j].q[1] = p[1];
        }
        #pragma unroll
        for (int i = 0; i < 2; ++i)
            #pragma unroll
            for (int j = 0; j < 4; ++j)
                acc[i][j] = __builtin_amdgcn_wmma_f32_16x16x32_bf16(
                    false, a[i].v, false, b[j].v, (short)0, acc[i][j], false,
                    false);
        __syncthreads();
    }

    // --- epilogue ---
    #pragma unroll
    for (int i = 0; i < 2; ++i) {
        #pragma unroll
        for (int j = 0; j < 4; ++j) {
            const int colg = bn + wn + j * 16 + l15;
            float bv = 0.f;
            if (BIAS) bv = bias[colg];
            #pragma unroll
            for (int r = 0; r < 8; ++r) {
                const int rowg = bm + wm + i * 16 + r + lhalf * 8;
                float v = acc[i][j][r] + bv;
                if (RELU) v = fmaxf(v, 0.f);
                if (RES) v += resid[(size_t)rowg * N + colg];
                if (OUTBF16)
                    ((unsigned short*)Cout)[(size_t)rowg * N + colg] =
                        f32_to_bf16(v);
                else
                    ((float*)Cout)[(size_t)rowg * N + colg] = v;
            }
        }
    }
}

// ---------------------------------------------------------------------------
// Flash attention (bf16 Q/K/V, bf16 ctx out). One wave handles one 16-query
// strip of one head. Per 32-key tile: QK^T = 4 WMMAs, online softmax with
// shfl_xor row reductions, P staged through per-wave LDS (C-layout ->
// A-layout), PV = 4 WMMAs. d_k = 64, scale = 0.125.
// ---------------------------------------------------------------------------
__global__ __launch_bounds__(256) void attention_kernel(
    const unsigned short* __restrict__ Q, const unsigned short* __restrict__ Kd,
    const unsigned short* __restrict__ V, unsigned short* __restrict__ O,
    int Bn, int H, int S, int DM) {
    __shared__ unsigned short Pbuf[8 * 16 * 32];  // 1KB per wave

    const int lane  = threadIdx.x & 31;
    const int wave  = threadIdx.x >> 5;
    const int l15   = lane & 15;
    const int lhalf = (lane < 16) ? 0 : 1;

    const int gw     = blockIdx.x * 8 + wave;
    const int strips = S / 16;
    const int b      = gw / (H * strips);
    const int rem    = gw % (H * strips);
    const int h      = rem / strips;
    const int qb     = (rem % strips) * 16;

    unsigned short* P = Pbuf + wave * 512;

    // Q A-fragments: rows qb..qb+15, d split per A 16x32 layout
    Frag aq[2];
    {
        const unsigned short* qp =
            Q + ((size_t)(b * S + qb + l15)) * DM + h * 64;
        #pragma unroll
        for (int f = 0; f < 2; ++f) {
            const int d0 = f * 32 + lhalf * 8;
            aq[f].q[0] = *(const u32x4*)(qp + d0);
            aq[f].q[1] = *(const u32x4*)(qp + d0 + 16);
        }
    }

    const v8f zero = {0.f, 0.f, 0.f, 0.f, 0.f, 0.f, 0.f, 0.f};
    float m[8], l[8];
    v8f o[4];
    #pragma unroll
    for (int r = 0; r < 8; ++r) { m[r] = -__builtin_inff(); l[r] = 0.f; }
    #pragma unroll
    for (int j = 0; j < 4; ++j) o[j] = zero;

    for (int kb = 0; kb < S; kb += 32) {
        // --- scores: two 16x16 C tiles over 32 keys ---
        v8f s[2];
        #pragma unroll
        for (int nt = 0; nt < 2; ++nt) {
            const unsigned short* kp =
                Kd + ((size_t)(b * S + kb + nt * 16 + l15)) * DM + h * 64;
            v8f c = zero;
            #pragma unroll
            for (int df = 0; df < 2; ++df) {
                Frag bk;
                const int d0 = df * 32 + lhalf * 8;  // B 32x16: table K order
                bk.q[0] = *(const u32x4*)(kp + d0);
                bk.q[1] = *(const u32x4*)(kp + d0 + 16);
                c = __builtin_amdgcn_wmma_f32_16x16x32_bf16(
                    false, aq[df].v, false, bk.v, (short)0, c, false, false);
            }
            s[nt] = c;
        }

        // --- online softmax (rows live across 16-lane groups) ---
        #pragma unroll
        for (int r = 0; r < 8; ++r) {
            float s0 = s[0][r] * 0.125f;
            float s1 = s[1][r] * 0.125f;
            float mx = fmaxf(s0, s1);
            #pragma unroll
            for (int off = 1; off < 16; off <<= 1)
                mx = fmaxf(mx, __shfl_xor(mx, off, 32));
            const float mnew  = fmaxf(m[r], mx);
            const float alpha = __expf(m[r] - mnew);
            const float p0 = __expf(s0 - mnew);
            const float p1 = __expf(s1 - mnew);
            float rs = p0 + p1;
            #pragma unroll
            for (int off = 1; off < 16; off <<= 1)
                rs += __shfl_xor(rs, off, 32);
            l[r] = l[r] * alpha + rs;
            m[r] = mnew;
            #pragma unroll
            for (int j = 0; j < 4; ++j) o[j][r] *= alpha;
            const int prow = r + lhalf * 8;
            P[prow * 32 + l15]      = f32_to_bf16(p0);
            P[prow * 32 + 16 + l15] = f32_to_bf16(p1);
        }

        // --- reload P as A fragment (16 rows x 32 keys) ---
        Frag pa;
        {
            const u32x4* pp = (const u32x4*)(P + l15 * 32);
            pa.q[0] = pp[lhalf * 2];
            pa.q[1] = pp[lhalf * 2 + 1];
        }

        // --- PV: V B-fragments (32 keys x 64 d) ---
        #pragma unroll
        for (int j = 0; j < 4; ++j) {
            Frag bv;
            const unsigned short* vp =
                V + ((size_t)(b * S + kb + lhalf * 16)) * DM + h * 64 +
                j * 16 + l15;
            #pragma unroll
            for (int kk = 0; kk < 16; ++kk) bv.u[kk] = vp[(size_t)kk * DM];
            o[j] = __builtin_amdgcn_wmma_f32_16x16x32_bf16(
                false, pa.v, false, bv.v, (short)0, o[j], false, false);
        }
    }

    // --- finalize: divide rows by l, store bf16 ctx [b,s,h*64+d] ---
    #pragma unroll
    for (int r = 0; r < 8; ++r) {
        const float inv = 1.f / l[r];
        unsigned short* op =
            O + ((size_t)(b * S + qb + r + lhalf * 8)) * DM + h * 64;
        #pragma unroll
        for (int j = 0; j < 4; ++j)
            op[j * 16 + l15] = f32_to_bf16(o[j][r] * inv);
    }
}

// ---------------------------------------------------------------------------
// Host: orchestrate the full pre-LN transformer layer.
// ---------------------------------------------------------------------------
extern "C" void kernel_launch(void* const* d_in, const int* in_sizes, int n_in,
                              void* d_out, int out_size, void* d_ws,
                              size_t ws_size, hipStream_t stream) {
    constexpr int Bn = 4, S = 2048, D = 1024, F = 4096, H = 16;
    constexpr int M = Bn * S;  // 8192 token rows

    const float* x    = (const float*)d_in[0];
    const float* W_q  = (const float*)d_in[1];
    const float* W_k  = (const float*)d_in[2];
    const float* W_v  = (const float*)d_in[3];
    const float* W_o  = (const float*)d_in[4];
    const float* W1   = (const float*)d_in[5];
    const float* b1   = (const float*)d_in[6];
    const float* W2   = (const float*)d_in[7];
    const float* b2   = (const float*)d_in[8];
    const float* g1   = (const float*)d_in[9];
    const float* be1  = (const float*)d_in[10];
    const float* g2   = (const float*)d_in[11];
    const float* be2  = (const float*)d_in[12];

    char* ws = (char*)d_ws;
    size_t off = 0;
    auto alloc = [&](size_t bytes) {
        void* p = ws + off;
        off += (bytes + 255) & ~(size_t)255;
        return p;
    };

    unsigned short* wq16 = (unsigned short*)alloc((size_t)D * D * 2);
    unsigned short* wk16 = (unsigned short*)alloc((size_t)D * D * 2);
    unsigned short* wv16 = (unsigned short*)alloc((size_t)D * D * 2);
    unsigned short* wo16 = (unsigned short*)alloc((size_t)D * D * 2);
    unsigned short* w116 = (unsigned short*)alloc((size_t)D * F * 2);
    unsigned short* w216 = (unsigned short*)alloc((size_t)F * D * 2);
    unsigned short* ln1  = (unsigned short*)alloc((size_t)M * D * 2);
    unsigned short* qb   = (unsigned short*)alloc((size_t)M * D * 2);
    unsigned short* kb   = (unsigned short*)alloc((size_t)M * D * 2);
    unsigned short* vb   = (unsigned short*)alloc((size_t)M * D * 2);
    unsigned short* ctx  = (unsigned short*)alloc((size_t)M * D * 2);
    float* attnres       = (float*)alloc((size_t)M * D * 4);
    unsigned short* ln2  = (unsigned short*)alloc((size_t)M * D * 2);
    unsigned short* hbuf = (unsigned short*)alloc((size_t)M * F * 2);

    // 1) weights -> bf16
    const int ndd = D * D, ndf = D * F;
    cvt_bf16_kernel<<<(ndd + 255) / 256, 256, 0, stream>>>(W_q, wq16, ndd);
    cvt_bf16_kernel<<<(ndd + 255) / 256, 256, 0, stream>>>(W_k, wk16, ndd);
    cvt_bf16_kernel<<<(ndd + 255) / 256, 256, 0, stream>>>(W_v, wv16, ndd);
    cvt_bf16_kernel<<<(ndd + 255) / 256, 256, 0, stream>>>(W_o, wo16, ndd);
    cvt_bf16_kernel<<<(ndf + 255) / 256, 256, 0, stream>>>(W1, w116, ndf);
    cvt_bf16_kernel<<<(ndf + 255) / 256, 256, 0, stream>>>(W2, w216, ndf);

    // 2) LN1
    layernorm_bf16_kernel<<<M, 256, 0, stream>>>(x, g1, be1, ln1, D);

    // 3) Q, K, V projections
    dim3 gDD(D / 128, M / 128);
    gemm_bf16_kernel<false, false, false, true><<<gDD, 256, 0, stream>>>(
        ln1, wq16, nullptr, nullptr, qb, M, D, D);
    gemm_bf16_kernel<false, false, false, true><<<gDD, 256, 0, stream>>>(
        ln1, wk16, nullptr, nullptr, kb, M, D, D);
    gemm_bf16_kernel<false, false, false, true><<<gDD, 256, 0, stream>>>(
        ln1, wv16, nullptr, nullptr, vb, M, D, D);

    // 4) attention
    attention_kernel<<<(Bn * H * (S / 16)) / 8, 256, 0, stream>>>(
        qb, kb, vb, ctx, Bn, H, S, D);

    // 5) output projection + residual (fp32)
    gemm_bf16_kernel<false, false, true, false><<<gDD, 256, 0, stream>>>(
        ctx, wo16, nullptr, x, attnres, M, D, D);

    // 6) LN2
    layernorm_bf16_kernel<<<M, 256, 0, stream>>>(attnres, g2, be2, ln2, D);

    // 7) FFN up: +b1, ReLU, bf16 out
    dim3 gDF(F / 128, M / 128);
    gemm_bf16_kernel<true, true, false, true><<<gDF, 256, 0, stream>>>(
        ln2, w116, b1, nullptr, hbuf, M, F, D);

    // 8) FFN down: +b2, +residual, fp32 out -> d_out
    gemm_bf16_kernel<true, false, true, false><<<gDD, 256, 0, stream>>>(
        hbuf, w216, b2, attnres, (float*)d_out, M, D, F);
}
